// ActorLstm_7756710936900
// MI455X (gfx1250) — compile-verified, matrix operand order checked by reading, not run on hardware
//
#include <hip/hip_runtime.h>

// Problem sizes (fixed by the reference).
#define T_   512
#define B_   128
#define OBS_ 512
#define H_   512
#define ACT_ 20
#define H4_  2048          // 4*H
#define K2_  1024          // concat K = H(x) + H(h)
#define LN_EPS 1e-6f

typedef __bf16 bf16;
typedef __attribute__((ext_vector_type(16))) __bf16 v16bf;
typedef __attribute__((ext_vector_type(8)))  __bf16 v8bf;
typedef __attribute__((ext_vector_type(8)))  float  v8f;

union FragBF { v16bf v; v8bf h[2]; };

// Build a 16x32 bf16 WMMA operand fragment for this lane.
// 16-bit A layout (ISA 7.12.2): lanes 0-15 hold K {k..k+7, k+16..k+23},
// lanes 16-31 hold K {k+8..k+15, k+24..k+31}. B (32x16) is the mirror with
// lane -> N, so storing W transposed [N][K] gives identical contiguous loads.
__device__ __forceinline__ v16bf make_frag(const bf16* lo, const bf16* hi) {
  FragBF f;
  f.h[0] = *reinterpret_cast<const v8bf*>(lo);
  f.h[1] = *reinterpret_cast<const v8bf*>(hi);
  return f.v;
}

__device__ __forceinline__ float sigm(float x) {
  return 1.0f / (1.0f + __expf(-x));
}
__device__ __forceinline__ float fast_tanh(float x) {
  x = fminf(fmaxf(x, -15.0f), 15.0f);
  float e = __expf(2.0f * x);
  return (e - 1.0f) / (e + 1.0f);
}

// ---------------------------------------------------------------------------
// One-time (per launch) weight prep: f32 [K][N] row-major -> bf16 [N][K].
// ---------------------------------------------------------------------------
__global__ void __launch_bounds__(256)
transpose_bf16_kernel(const float* __restrict__ W, bf16* __restrict__ Wt,
                      int K, int N) {
  int idx = blockIdx.x * 256 + threadIdx.x;
  if (idx >= K * N) return;
  int n = idx / K, k = idx - n * K;                 // writes coalesced in k
  Wt[idx] = (bf16)W[(size_t)k * N + n];
}

// Wcat[n][k] (bf16, [2048][1024]): k<512 -> Wi[k][n], else Wh[k-512][n].
__global__ void __launch_bounds__(256)
build_wcat_kernel(const float* __restrict__ Wi, const float* __restrict__ Wh,
                  bf16* __restrict__ Wcat) {
  int idx = blockIdx.x * 256 + threadIdx.x;        // over 2048*1024
  int n = idx >> 10, k = idx & 1023;
  float v = (k < H_) ? Wi[(size_t)k * H4_ + n]
                     : Wh[(size_t)(k - H_) * H4_ + n];
  Wcat[idx] = (bf16)v;
}

__global__ void __launch_bounds__(256)
init_state_kernel(const float* __restrict__ cc, const float* __restrict__ ch,
                  float* __restrict__ c0, float* __restrict__ h0,
                  float* __restrict__ c1, float* __restrict__ h1) {
  int idx = blockIdx.x * 256 + threadIdx.x;        // over B*H
  if (idx >= B_ * H_) return;
  c0[idx] = cc[idx];            c1[idx] = cc[B_ * H_ + idx];
  h0[idx] = ch[idx];            h1[idx] = ch[B_ * H_ + idx];
}

__global__ void __launch_bounds__(256)
write_carry_kernel(const float* __restrict__ c0, const float* __restrict__ c1,
                   const float* __restrict__ h0, const float* __restrict__ h1,
                   float* __restrict__ out_c, float* __restrict__ out_h) {
  int idx = blockIdx.x * 256 + threadIdx.x;
  if (idx >= B_ * H_) return;
  out_c[idx] = c0[idx];  out_c[B_ * H_ + idx] = c1[idx];
  out_h[idx] = h0[idx];  out_h[B_ * H_ + idx] = h1[idx];
}

// ---------------------------------------------------------------------------
// Fused Dense(OBS->H) + LayerNorm + ReLU, bf16 out. One WG = 16 rows, all H.
// ---------------------------------------------------------------------------
__global__ void __launch_bounds__(256)
pre_ln_relu_kernel(const float* __restrict__ x, const bf16* __restrict__ Wt,
                   const float* __restrict__ bias, const float* __restrict__ ln_s,
                   const float* __restrict__ ln_b, bf16* __restrict__ out) {
  __shared__ __align__(16) bf16 Alds[16][OBS_];   // 16 KB
  __shared__ float Zlds[16][H_];                  // 32 KB
  __shared__ float red1[16][16], red2[16][16];
  __shared__ float sh_mu[16], sh_rs[16];

  const int  tid = threadIdx.x;
  const long rowbase = (long)blockIdx.x * 16;

  // Stage A: 16x512 f32 -> bf16.
  #pragma unroll
  for (int i = 0; i < 32; ++i) {
    int idx = i * 256 + tid;
    int r = idx >> 9, c = idx & 511;
    Alds[r][c] = (bf16)x[(rowbase + r) * OBS_ + c];
  }
  __syncthreads();

  const int lane = tid & 31, wave = tid >> 5;
  const int half = lane >> 4, m = lane & 15;
  const bf16* Arow = &Alds[m][0];

  for (int nt = 0; nt < 4; ++nt) {
    const int n0 = wave * 64 + nt * 16;
    const bf16* Brow = Wt + (size_t)(n0 + m) * OBS_;
    v8f acc = {};
    for (int k = 0; k < OBS_; k += 32) {
      v16bf a = make_frag(Arow + k + half * 8, Arow + k + 16 + half * 8);
      v16bf b = make_frag(Brow + k + half * 8, Brow + k + 16 + half * 8);
      acc = __builtin_amdgcn_wmma_f32_16x16x32_bf16(false, a, false, b,
                                                    (short)0, acc, false, false);
    }
    #pragma unroll
    for (int g = 0; g < 8; ++g)
      Zlds[g + 8 * half][n0 + m] = acc[g];
  }
  __syncthreads();

  // LayerNorm over the 512-wide row, then ReLU, then bf16 store.
  const int r = tid >> 4, sub = tid & 15;
  float s1 = 0.0f, s2 = 0.0f;
  for (int j = 0; j < 32; ++j) {
    int c = sub * 32 + j;
    float z = Zlds[r][c] + bias[c];
    s1 += z;  s2 += z * z;
  }
  red1[r][sub] = s1;  red2[r][sub] = s2;
  __syncthreads();
  if (sub == 0) {
    float t1 = 0.0f, t2 = 0.0f;
    for (int j = 0; j < 16; ++j) { t1 += red1[r][j]; t2 += red2[r][j]; }
    float mu  = t1 * (1.0f / H_);
    float var = t2 * (1.0f / H_) - mu * mu;
    sh_mu[r] = mu;
    sh_rs[r] = rsqrtf(var + LN_EPS);
  }
  __syncthreads();
  const float mu = sh_mu[r], rs = sh_rs[r];
  for (int j = 0; j < 32; ++j) {
    int c = sub * 32 + j;
    float z = Zlds[r][c] + bias[c];
    float v = fmaxf((z - mu) * rs * ln_s[c] + ln_b[c], 0.0f);
    out[(rowbase + r) * H_ + c] = (bf16)v;
  }
}

// ---------------------------------------------------------------------------
// One LSTM timestep. Grid = 32 WGs: blockIdx&7 -> 16-row block,
// blockIdx>>3 -> 128-column block of each gate section. Each wave owns one
// 16-col gate group with 4 accumulators (i,f,g,o) and gates in registers.
// z = [x_t , keep*h_{t-1}] @ Wcat  (K = 1024, bf16 WMMA, f32 accumulate).
// ---------------------------------------------------------------------------
__global__ void __launch_bounds__(256)
lstm_step_kernel(const bf16* __restrict__ x_t, const int* __restrict__ resets_t,
                 const bf16* __restrict__ Wcat, const float* __restrict__ bias,
                 float* __restrict__ c_state, float* __restrict__ h_state,
                 bf16* __restrict__ h_seq_t, float* __restrict__ h_f32_t) {
  __shared__ __align__(16) bf16 Alds[16][K2_];    // 32 KB
  __shared__ float keep[16];

  const int tid     = threadIdx.x;
  const int rowbase = (blockIdx.x & 7) * 16;
  const int colblk  = blockIdx.x >> 3;            // 0..3

  if (tid < 16) keep[tid] = (resets_t[rowbase + tid] != 0) ? 0.0f : 1.0f;

  // Stage x_t (already bf16) as raw u32 pairs.
  {
    const unsigned int* src = reinterpret_cast<const unsigned int*>(x_t);
    #pragma unroll
    for (int i = 0; i < 16; ++i) {
      int idx = i * 256 + tid;                    // 4096 u32 = 16x512 bf16
      int r = idx >> 8, c2 = idx & 255;
      reinterpret_cast<unsigned int*>(&Alds[r][0])[c2] =
          src[(size_t)(rowbase + r) * (H_ / 2) + c2];
    }
  }
  __syncthreads();                                // keep[] visible
  // Stage keep*h_{t-1} as bf16 into the second K half.
  #pragma unroll
  for (int i = 0; i < 32; ++i) {
    int idx = i * 256 + tid;
    int r = idx >> 9, c = idx & 511;
    float hv = h_state[(size_t)(rowbase + r) * H_ + c] * keep[r];
    Alds[r][H_ + c] = (bf16)hv;
  }
  __syncthreads();

  const int lane = tid & 31, wave = tid >> 5;
  const int half = lane >> 4, m = lane & 15;
  const bf16* Arow = &Alds[m][0];

  const int n0 = colblk * 128 + wave * 16;        // gate-group column base
  const bf16* B0 = Wcat + (size_t)(0 * H_ + n0 + m) * K2_;
  const bf16* B1 = Wcat + (size_t)(1 * H_ + n0 + m) * K2_;
  const bf16* B2 = Wcat + (size_t)(2 * H_ + n0 + m) * K2_;
  const bf16* B3 = Wcat + (size_t)(3 * H_ + n0 + m) * K2_;

  v8f ai = {}, af = {}, ag = {}, ao = {};
  for (int k = 0; k < K2_; k += 32) {
    v16bf a  = make_frag(Arow + k + half * 8, Arow + k + 16 + half * 8);
    v16bf b0 = make_frag(B0 + k + half * 8, B0 + k + 16 + half * 8);
    ai = __builtin_amdgcn_wmma_f32_16x16x32_bf16(false, a, false, b0, (short)0, ai, false, false);
    v16bf b1 = make_frag(B1 + k + half * 8, B1 + k + 16 + half * 8);
    af = __builtin_amdgcn_wmma_f32_16x16x32_bf16(false, a, false, b1, (short)0, af, false, false);
    v16bf b2 = make_frag(B2 + k + half * 8, B2 + k + 16 + half * 8);
    ag = __builtin_amdgcn_wmma_f32_16x16x32_bf16(false, a, false, b2, (short)0, ag, false, false);
    v16bf b3 = make_frag(B3 + k + half * 8, B3 + k + 16 + half * 8);
    ao = __builtin_amdgcn_wmma_f32_16x16x32_bf16(false, a, false, b3, (short)0, ao, false, false);
  }

  // Gate in registers: accumulator element g is row (g + 8*half), col n0+m.
  const int ncol = n0 + m;
  const float bi = bias[ncol],          bf_ = bias[H_ + ncol];
  const float bg = bias[2 * H_ + ncol], bo  = bias[3 * H_ + ncol];
  #pragma unroll
  for (int g = 0; g < 8; ++g) {
    const int r = g + 8 * half;
    const size_t off = (size_t)(rowbase + r) * H_ + ncol;
    float iv = sigm(ai[g] + bi);
    float fv = sigm(af[g] + bf_);
    float gv = fast_tanh(ag[g] + bg);
    float ov = sigm(ao[g] + bo);
    float nc = fv * (c_state[off] * keep[r]) + iv * gv;
    float nh = ov * fast_tanh(nc);
    c_state[off] = nc;
    h_state[off] = nh;
    if (h_seq_t) h_seq_t[off] = (bf16)nh;   // feeds next layer (bf16)
    if (h_f32_t) h_f32_t[off] = nh;         // embedding output (f32)
  }
}

// ---------------------------------------------------------------------------
// Fused Dense(H->H) + bias + ReLU, bf16 out (feeds the 20-wide head).
// ---------------------------------------------------------------------------
__global__ void __launch_bounds__(256)
post_relu_kernel(const float* __restrict__ emb, const bf16* __restrict__ Wt,
                 const float* __restrict__ bias, bf16* __restrict__ p_out) {
  __shared__ __align__(16) bf16 Alds[16][H_];
  const int  tid = threadIdx.x;
  const long rowbase = (long)blockIdx.x * 16;

  #pragma unroll
  for (int i = 0; i < 32; ++i) {
    int idx = i * 256 + tid;
    int r = idx >> 9, c = idx & 511;
    Alds[r][c] = (bf16)emb[(rowbase + r) * H_ + c];
  }
  __syncthreads();

  const int lane = tid & 31, wave = tid >> 5;
  const int half = lane >> 4, m = lane & 15;
  const bf16* Arow = &Alds[m][0];

  for (int nt = 0; nt < 4; ++nt) {
    const int n0 = wave * 64 + nt * 16;
    const bf16* Brow = Wt + (size_t)(n0 + m) * H_;
    v8f acc = {};
    for (int k = 0; k < H_; k += 32) {
      v16bf a = make_frag(Arow + k + half * 8, Arow + k + 16 + half * 8);
      v16bf b = make_frag(Brow + k + half * 8, Brow + k + 16 + half * 8);
      acc = __builtin_amdgcn_wmma_f32_16x16x32_bf16(false, a, false, b,
                                                    (short)0, acc, false, false);
    }
    const int ncol = n0 + m;
    #pragma unroll
    for (int g = 0; g < 8; ++g) {
      float v = fmaxf(acc[g] + bias[ncol], 0.0f);
      p_out[(rowbase + g + 8 * half) * H_ + ncol] = (bf16)v;
    }
  }
}

// logits = p @ out_W + out_b  (N=20; tiny, VALU dot products).
__global__ void __launch_bounds__(256)
out_head_kernel(const bf16* __restrict__ p, const float* __restrict__ W,
                const float* __restrict__ b, float* __restrict__ logits) {
  int idx = blockIdx.x * 256 + threadIdx.x;       // over T*B*ACT
  if (idx >= T_ * B_ * ACT_) return;
  int row = idx / ACT_, a = idx - row * ACT_;
  const bf16* pr = p + (size_t)row * H_;
  float s = b[a];
  for (int k = 0; k < H_; ++k)
    s += (float)pr[k] * W[k * ACT_ + a];
  logits[idx] = s;
}

// ---------------------------------------------------------------------------
extern "C" void kernel_launch(void* const* d_in, const int* in_sizes, int n_in,
                              void* d_out, int out_size, void* d_ws, size_t ws_size,
                              hipStream_t stream) {
  (void)in_sizes; (void)n_in; (void)out_size;

  const float* x       = (const float*)d_in[0];
  const int*   resets  = (const int*)  d_in[1];
  const float* carry_c = (const float*)d_in[2];
  const float* carry_h = (const float*)d_in[3];
  const float* pre_W   = (const float*)d_in[4];
  const float* pre_b   = (const float*)d_in[5];
  const float* ln_s    = (const float*)d_in[6];
  const float* ln_b    = (const float*)d_in[7];
  const float* Wi0     = (const float*)d_in[8];
  const float* Wh0     = (const float*)d_in[9];
  const float* b0      = (const float*)d_in[10];
  const float* Wi1     = (const float*)d_in[11];
  const float* Wh1     = (const float*)d_in[12];
  const float* b1      = (const float*)d_in[13];
  const float* post_W  = (const float*)d_in[14];
  const float* post_b  = (const float*)d_in[15];
  const float* out_W   = (const float*)d_in[16];
  const float* out_b   = (const float*)d_in[17];

  // d_out layout: new_c [2,B,H] | new_h [2,B,H] | logits [T,B,ACT] | emb [T,B,H]
  const size_t BH = (size_t)B_ * H_;
  float* out     = (float*)d_out;
  float* out_c   = out;
  float* out_h   = out + 2 * BH;
  float* out_log = out + 4 * BH;
  float* out_emb = out_log + (size_t)T_ * B_ * ACT_;

  // Workspace carve-out (~138 MB).
  char*  ws  = (char*)d_ws;
  size_t off = 0;
  auto carve = [&](size_t bytes) -> char* {
    char* p = ws + off;
    off += (bytes + 255) & ~(size_t)255;
    return p;
  };
  bf16*  pre_seq = (bf16*)carve((size_t)T_ * BH * sizeof(bf16)); // reused as p
  bf16*  h0_seq  = (bf16*)carve((size_t)T_ * BH * sizeof(bf16));
  bf16*  Wcat0   = (bf16*)carve((size_t)H4_ * K2_ * sizeof(bf16));
  bf16*  Wcat1   = (bf16*)carve((size_t)H4_ * K2_ * sizeof(bf16));
  bf16*  preWt   = (bf16*)carve((size_t)H_ * OBS_ * sizeof(bf16));
  bf16*  postWt  = (bf16*)carve((size_t)H_ * H_ * sizeof(bf16));
  float* c0 = (float*)carve(BH * sizeof(float));
  float* h0 = (float*)carve(BH * sizeof(float));
  float* c1 = (float*)carve(BH * sizeof(float));
  float* h1 = (float*)carve(BH * sizeof(float));
  if (off > ws_size) return;   // fixed sizes; should never trigger

  // 1) State init + weight conversion (all deterministic, re-run each replay).
  init_state_kernel<<<(B_ * H_ + 255) / 256, 256, 0, stream>>>(carry_c, carry_h,
                                                              c0, h0, c1, h1);
  transpose_bf16_kernel<<<(OBS_ * H_ + 255) / 256, 256, 0, stream>>>(pre_W, preWt, OBS_, H_);
  transpose_bf16_kernel<<<(H_ * H_ + 255) / 256, 256, 0, stream>>>(post_W, postWt, H_, H_);
  build_wcat_kernel<<<(H4_ * K2_) / 256, 256, 0, stream>>>(Wi0, Wh0, Wcat0);
  build_wcat_kernel<<<(H4_ * K2_) / 256, 256, 0, stream>>>(Wi1, Wh1, Wcat1);

  // 2) Front-end: Dense + LN + ReLU over all T*B rows.
  pre_ln_relu_kernel<<<(T_ * B_) / 16, 256, 0, stream>>>(x, preWt, pre_b,
                                                         ln_s, ln_b, pre_seq);

  // 3) Sequential recurrence: one launch per (layer, t), stream-ordered.
  for (int t = 0; t < T_; ++t) {
    lstm_step_kernel<<<32, 256, 0, stream>>>(
        pre_seq + (size_t)t * BH, resets + (size_t)t * B_, Wcat0, b0,
        c0, h0, h0_seq + (size_t)t * BH, (float*)nullptr);
    lstm_step_kernel<<<32, 256, 0, stream>>>(
        h0_seq + (size_t)t * BH, resets + (size_t)t * B_, Wcat1, b1,
        c1, h1, (bf16*)nullptr, out_emb + (size_t)t * BH);
  }

  // 4) Head: Dense+ReLU (WMMA) then H->20 projection.
  bf16* p_seq = pre_seq;                       // reuse front-end buffer
  post_relu_kernel<<<(T_ * B_) / 16, 256, 0, stream>>>(out_emb, postWt, post_b, p_seq);
  out_head_kernel<<<(T_ * B_ * ACT_ + 255) / 256, 256, 0, stream>>>(p_seq, out_W,
                                                                    out_b, out_log);

  // 5) Final carries.
  write_carry_kernel<<<(B_ * H_ + 255) / 256, 256, 0, stream>>>(c0, c1, h0, h1,
                                                                out_c, out_h);
}
